// SemanticsModulatedAttention_79731772883426
// MI455X (gfx1250) — matrix operand order repaired and checked by reading, not compiled, not run on hardware
//
#include <hip/hip_runtime.h>

// ---------------------------------------------------------------------------
// SemanticsModulatedAttention for MI455X (gfx1250), wave32 + WMMA.
//
// Input flattening assumption: setup_inputs() dict in insertion order, nested
// params dict recursively in insertion order:
//   0 x, 1 text_emb, 2 joint_emb, 3 other_emb, 4 src_mask, 5 joint_mask,
//   6 norm_g, 7 norm_b, 8 q_w, 9 q_b,
//   10..23 text{norm_x_g,norm_x_b,norm_c_g,norm_c_b,kc_w,kc_b,vc_w,vc_b,
//               kx_w,kx_b,vx_w,vx_b,proj_w,proj_b}
//   24..37 joint{...same...}
//   38..45 y{norm_x_g,norm_x_b,kx_w,kx_b,vx_w,vx_b,proj_w,proj_b}
//   46..51 style{emb_w,emb_b,norm_g,norm_b,out_w,out_b}
//   52 n_text, 53 n_both, 54 n_stick (device int scalars)
// ---------------------------------------------------------------------------

typedef __attribute__((ext_vector_type(16))) _Float16 v16h;
typedef __attribute__((ext_vector_type(8)))  _Float16 v8h;
typedef __attribute__((ext_vector_type(4)))  _Float16 v4h;
typedef __attribute__((ext_vector_type(8)))  float    v8f;

#define H_   8
#define D_   256
#define T_   1024
#define B_   64
#define NT_  77
#define NJ_  32
#define TE_  1024
#define EPS_ 1e-5f

// weight slots in the swizzled-f16 weight arena (each slot 65536 halves)
enum { SL_Q=0, SL_TKX, SL_TVX, SL_TKC, SL_TVC, SL_TPROJ,
       SL_JKX, SL_JVX, SL_JKC, SL_JVC, SL_JPROJ,
       SL_YKX, SL_YVX, SL_YPROJ, SL_OUT, NSLOT };

__device__ inline v8f wmma_(v16h a, v16h b, v8f c) {
  return __builtin_amdgcn_wmma_f32_16x16x32_f16(
      false, a, false, b, (short)0, c, false, false);
}

// ---------------------------------------------------------------------------
// Kernel 0: fuse LN affine into weights; emit f16 B-fragment-swizzled weights
// W'[o][c] = W[o][c]*g[c]  stored at [ct][kt][lane][e] with
// lane=(o&15)+16*(((c&31)>>3)&1), e=(c&7)+8*((c&31)>>4) ;  fused bias fb[o]=b[o]+sum_c W[o][c]*bv[c]
// ---------------------------------------------------------------------------
struct FuseDesc { const float* W; const float* g; const float* bv; const float* b; int Dc; };
struct FuseArgs { FuseDesc d[NSLOT]; };

__global__ __launch_bounds__(256) void fuse_weights(FuseArgs fa,
    _Float16* __restrict__ WS, float* __restrict__ FB)
{
  int id = blockIdx.x, tid = threadIdx.x;
  FuseDesc dd = fa.d[id];
  int Dc = dd.Dc;
  _Float16* ow = WS + (size_t)id * 65536;
  for (int idx = tid; idx < 256 * Dc; idx += 256) {
    int o = idx / Dc, c = idx - o * Dc;
    float wv = dd.W[idx];
    if (dd.g) wv *= dd.g[c];
    int ct = o >> 4, kt = c >> 5, kl = c & 31;
    int lane = (o & 15) + 16 * ((kl >> 3) & 1);
    ow[(((size_t)ct * (Dc >> 5) + kt) * 32 + lane) * 16 + (kl & 7) + 8 * (kl >> 4)] = (_Float16)wv;
  }
  int o = tid;  // Do == 256 for all slots
  float a = dd.b ? dd.b[o] : 0.f;
  if (dd.bv) {
    const float* wr = dd.W + (size_t)o * Dc;
    for (int c = 0; c < Dc; ++c) a += wr[c] * dd.bv[c];
  }
  FB[id * 256 + o] = a;
}

// ---------------------------------------------------------------------------
// Kernel 1: per-row LayerNorm (no affine) -> f16, stored in A-fragment layout
// out tile base = (b*tilesPerB + rt)*(C/32)*512 ; pad rows (>=R) written as 0.
// ---------------------------------------------------------------------------
__global__ __launch_bounds__(256) void prep_ln(const float* __restrict__ src,
    _Float16* __restrict__ out, int R, int tilesPerB, int C)
{
  int b = blockIdx.x, rt = blockIdx.y, tid = threadIdx.x;
  int m = tid >> 4, cs = tid & 15;
  int row = rt * 16 + m;
  int cp = C >> 4;
  __shared__ float red[2][16][16];
  __shared__ float mrs[16][2];
  float vals[16];
  float s = 0.f, s2 = 0.f;
  bool valid = row < R;
  const float* sp = src + ((size_t)b * R + (valid ? row : 0)) * C + cs * cp;
  for (int j = 0; j < cp; ++j) {
    float v = valid ? sp[j] : 0.f;
    vals[j] = v; s += v; s2 += v * v;
  }
  red[0][m][cs] = s; red[1][m][cs] = s2;
  __syncthreads();
  if (cs == 0) {
    float a = 0.f, a2 = 0.f;
    for (int i = 0; i < 16; ++i) { a += red[0][m][i]; a2 += red[1][m][i]; }
    float mean = a / (float)C;
    float var = a2 / (float)C - mean * mean;
    mrs[m][0] = mean; mrs[m][1] = rsqrtf(var + EPS_);
  }
  __syncthreads();
  float mean = mrs[m][0], rstd = mrs[m][1];
  _Float16* ot = out + (size_t)(b * tilesPerB + rt) * (C >> 5) * 512;
  if (C == 256) {
    v8h lo, hi;
    for (int j = 0; j < 8; ++j) {
      lo[j] = (_Float16)(valid ? (vals[j]     - mean) * rstd : 0.f);
      hi[j] = (_Float16)(valid ? (vals[8 + j] - mean) * rstd : 0.f);
    }
    int kt = cs >> 1, eo = 8 * (cs & 1);
    *(v8h*)&ot[kt * 512 + m * 16 + eo]        = lo;   // j<8  -> lane m
    *(v8h*)&ot[kt * 512 + (m + 16) * 16 + eo] = hi;   // j>=8 -> lane m+16
  } else { // C == 64
    v4h q;
    for (int j = 0; j < 4; ++j)
      q[j] = (_Float16)(valid ? (vals[j] - mean) * rstd : 0.f);
    int kt = cs >> 3;
    int lane = m + 16 * ((cs >> 1) & 1);
    int eo = 8 * ((cs >> 2) & 1) + 4 * (cs & 1);
    *(v4h*)&ot[kt * 512 + lane * 16 + eo] = q;
  }
}

// ---------------------------------------------------------------------------
// Kernel 2: query = xhat @ Wq'^T + bq'  -> q_orig and qfin (row-major f32)
// grid (B, T/16); 8 waves, wave w owns cols [32w,32w+32)
// ---------------------------------------------------------------------------
__global__ __launch_bounds__(256) void q_gemm(const _Float16* __restrict__ xA,
    const _Float16* __restrict__ WS, const float* __restrict__ FB,
    float* __restrict__ q0, float* __restrict__ qf)
{
  int b = blockIdx.x, rt = blockIdx.y, tid = threadIdx.x;
  int w = tid >> 5, lane = tid & 31, l4 = lane & 15, g = lane >> 4;
  const _Float16* A = xA + (size_t)(b * 64 + rt) * 8 * 512;
  v8f acc[2] = {};
  for (int kt = 0; kt < 8; ++kt) {
    v16h a = *(const v16h*)(A + kt * 512 + lane * 16);
    for (int cl = 0; cl < 2; ++cl) {
      int ct = 2 * w + cl;
      v16h bw = *(const v16h*)(WS + (size_t)SL_Q * 65536 + ((ct * 8 + kt) * 32 + lane) * 16);
      acc[cl] = wmma_(a, bw, acc[cl]);
    }
  }
  for (int cl = 0; cl < 2; ++cl) {
    int col = 32 * w + 16 * cl + l4;
    float bias = FB[SL_Q * 256 + col];
    for (int r = 0; r < 8; ++r) {
      size_t addr = ((size_t)b * T_ + rt * 16 + r + 8 * g) * 256 + col;
      float v = acc[cl][r] + bias;
      q0[addr] = v; qf[addr] = v;
    }
  }
}

// ---------------------------------------------------------------------------
// Kernel 3: linear-attention builder. grid (B, branch=3, seg=5), wave = head.
// Per 32-row chunk: K,V via WMMA; P=exp(K+b+mask); P,V swizzled into per-wave
// LDS A/B fragments; acc += P^T V via WMMA (no running max: |key| <= ~2).
// Writes per-seg partial acc (f32 [d][l]) and partial col-sums.
// ---------------------------------------------------------------------------
__global__ __launch_bounds__(256) void attn_build(
    const _Float16* __restrict__ xA, const _Float16* __restrict__ tcA,
    const _Float16* __restrict__ jcA, const _Float16* __restrict__ WS,
    const float* __restrict__ FB, const float* __restrict__ smask,
    const float* __restrict__ jmask,
    const int* __restrict__ pnt, const int* __restrict__ pnb, const int* __restrict__ pns,
    float* __restrict__ patt, float* __restrict__ pS)
{
  int b = blockIdx.x, br = blockIdx.y, seg = blockIdx.z;
  int c1 = *pnt, c2 = c1 + *pnb, c3 = c2 + *pns;
  int condTiles, condKT, kxS, vxS, kcS = 0, vcS = 0;
  const _Float16* cA = nullptr;
  if (br == 0) {
    if (b >= c2) return;
    condTiles = 5; condKT = 8; cA = tcA + (size_t)b * 5 * 8 * 512;
    kxS = SL_TKX; vxS = SL_TVX; kcS = SL_TKC; vcS = SL_TVC;
  } else if (br == 1) {
    if (b < c1 || b >= c3) return;
    condTiles = 2; condKT = 2; cA = jcA + (size_t)b * 2 * 2 * 512;
    kxS = SL_JKX; vxS = SL_JVX; kcS = SL_JKC; vcS = SL_JVC;
  } else {
    condTiles = 0; condKT = 0; kxS = SL_YKX; vxS = SL_YVX;
  }
  int tot = condTiles + 64;
  int t0 = seg * 16;
  if (t0 >= tot) return;
  int t1 = t0 + 16 < tot ? t0 + 16 : tot;

  __shared__ alignas(32) _Float16 ldsP[8][2][32][16]; // [wave][dt][lane][e]
  __shared__ alignas(32) _Float16 ldsV[8][2][32][16]; // [wave][lt][lane][e]

  int tid = threadIdx.x, w = tid >> 5, lane = tid & 31, l4 = lane & 15, g = lane >> 4;
  v8f acc[2][2] = {};
  float Sp0 = 0.f, Sp1 = 0.f;

  for (int tc0 = t0; tc0 < t1; tc0 += 2) {
    int nt = (t1 - tc0 >= 2) ? 2 : 1;
    for (int sub = 0; sub < nt; ++sub) {
      int t = tc0 + sub;
      bool isC = t < condTiles;
      const _Float16* A; int nkt, ks, vs;
      if (isC) { A = cA + (size_t)t * condKT * 512; nkt = condKT; ks = kcS; vs = vcS; }
      else     { A = xA + (size_t)(b * 64 + (t - condTiles)) * 8 * 512; nkt = 8; ks = kxS; vs = vxS; }
      for (int cl = 0; cl < 2; ++cl) {
        int ct = 2 * w + cl;
        v8f kf = {}, vf = {};
        for (int kt = 0; kt < nkt; ++kt) {
          v16h a  = *(const v16h*)(A + kt * 512 + lane * 16);
          v16h bk = *(const v16h*)(WS + (size_t)ks * 65536 + ((ct * nkt + kt) * 32 + lane) * 16);
          v16h bv = *(const v16h*)(WS + (size_t)vs * 65536 + ((ct * nkt + kt) * 32 + lane) * 16);
          kf = wmma_(a, bk, kf);
          vf = wmma_(a, bv, vf);
        }
        int col = 32 * w + 16 * cl + l4;
        float kb = FB[ks * 256 + col], vb = FB[vs * 256 + col];
        v8h pk, vk;
        float sp = 0.f;
        for (int r = 0; r < 8; ++r) {
          int mrow = r + 8 * g;
          float mval;
          if (!isC)        mval = smask[(size_t)b * T_ + (t - condTiles) * 16 + mrow];
          else if (br == 0) mval = (t * 16 + mrow < NT_) ? 1.f : 0.f;
          else { int n = t * 16 + mrow; mval = (n < NJ_) ? jmask[(size_t)b * NJ_ + n] : 0.f; }
          float kv = kf[r] + kb - (1.f - mval) * 1000000.f;
          float p = __expf(kv);
          sp += p;
          pk[r] = (_Float16)p;
          vk[r] = (_Float16)((vf[r] + vb) * mval);
        }
        if (cl == 0) Sp0 += sp; else Sp1 += sp;
        // P^T A-frag: (d=16cl+l4, k=16sub+r+8g) -> lane'=lane, e=8sub+r
        *(v8h*)&ldsP[w][cl][lane][8 * sub] = pk;
        // V B-frag: (k=16sub+r+8g, n=16cl+l4) -> lane'=l4+16g, e=8sub+r
        *(v8h*)&ldsV[w][cl][l4 + 16 * g][8 * sub] = vk;
      }
    }
    if (nt == 1) { // pad missing second 16-row half with zeros
      v8h z = {};
      *(v8h*)&ldsP[w][0][lane][8] = z;
      *(v8h*)&ldsP[w][1][lane][8] = z;
      *(v8h*)&ldsV[w][0][l4 + 16 * g][8] = z;
      *(v8h*)&ldsV[w][1][l4 + 16 * g][8] = z;
    }
    for (int dt = 0; dt < 2; ++dt) {
      v16h aP = *(const v16h*)&ldsP[w][dt][lane][0];
      for (int lt = 0; lt < 2; ++lt) {
        v16h bV = *(const v16h*)&ldsV[w][lt][lane][0];
        acc[dt][lt] = wmma_(aP, bV, acc[dt][lt]);
      }
    }
  }
  size_t pb = ((((size_t)b * 3 + br) * 5 + seg) * 8 + w) * 1024;
  for (int dt = 0; dt < 2; ++dt)
    for (int lt = 0; lt < 2; ++lt)
      for (int r = 0; r < 8; ++r) {
        int d = 16 * dt + r + 8 * g, l = 16 * lt + l4;
        patt[pb + d * 32 + l] = acc[dt][lt][r];
      }
  size_t sb = ((((size_t)b * 3 + br) * 5 + seg) * 8 + w) * 64;
  pS[sb + lane]      = Sp0;
  pS[sb + 32 + lane] = Sp1;
}

// ---------------------------------------------------------------------------
// Kernel 4: reduce per-seg partials, divide by column sums, emit att as f16
// B-fragments [lt][lane][e] per (b,branch,head). grid (B,3).
// ---------------------------------------------------------------------------
__global__ __launch_bounds__(256) void finalize_att(
    const float* __restrict__ patt, const float* __restrict__ pS,
    _Float16* __restrict__ attB,
    const int* __restrict__ pnt, const int* __restrict__ pnb, const int* __restrict__ pns)
{
  int b = blockIdx.x, br = blockIdx.y;
  int c1 = *pnt, c2 = c1 + *pnb, c3 = c2 + *pns;
  if (br == 0 && b >= c2) return;
  if (br == 1 && (b < c1 || b >= c3)) return;
  int nseg = (br == 2) ? 4 : 5;
  int tid = threadIdx.x, w = tid >> 5, lane = tid & 31; // lane == d
  int cl = lane >> 4, dl = lane & 15;
  float s = 0.f;
  for (int seg = 0; seg < nseg; ++seg) {
    size_t sb = ((((size_t)b * 3 + br) * 5 + seg) * 8 + w) * 64;
    s += pS[sb + cl * 32 + dl] + pS[sb + cl * 32 + dl + 16];
  }
  float inv = 1.f / s;
  _Float16* ob = attB + (((size_t)b * 3 + br) * 8 + w) * 1024;
  for (int l = 0; l < 32; ++l) {
    float a = 0.f;
    for (int seg = 0; seg < nseg; ++seg) {
      size_t pb = ((((size_t)b * 3 + br) * 5 + seg) * 8 + w) * 1024;
      a += patt[pb + lane * 32 + l];
    }
    int lt = l >> 4, lanep = (l & 15) + 16 * ((lane >> 3) & 1), e = (lane & 7) + 8 * (lane >> 4);
    ob[(lt * 32 + lanep) * 16 + e] = (_Float16)(a * inv);
  }
}

// ---------------------------------------------------------------------------
// Kernel 5: y = softmax_d(q) @ att ; z = y @ proj^T + proj_b ;
// mode 0: outb += z (query update), mode 1: outb = z (q2). grid (B, T/16).
// ---------------------------------------------------------------------------
__global__ __launch_bounds__(256) void y_apply(
    const float* __restrict__ qin, const _Float16* __restrict__ attB,
    const _Float16* __restrict__ WS, const float* __restrict__ FB,
    int projSlot, int br, int mode, float* __restrict__ outb,
    const int* __restrict__ pnt, const int* __restrict__ pnb, const int* __restrict__ pns)
{
  int b = blockIdx.x, rt = blockIdx.y;
  int c1 = *pnt, c2 = c1 + *pnb, c3 = c2 + *pns;
  if (br == 0 && b >= c2) return;
  if (br == 1 && (b < c1 || b >= c3)) return;
  __shared__ alignas(32) _Float16 sfrag[8][32][16];
  int tid = threadIdx.x;
  if (tid < 128) { // (row m, head h) softmax over 32 dims -> A-frag in LDS
    int m = tid >> 3, h = tid & 7;
    const float* qp = qin + ((size_t)b * T_ + rt * 16 + m) * 256 + 32 * h;
    float v[32];
    float mx = -1e30f;
    for (int j = 0; j < 32; ++j) { v[j] = qp[j]; mx = fmaxf(mx, v[j]); }
    float s = 0.f;
    for (int j = 0; j < 32; ++j) { v[j] = __expf(v[j] - mx); s += v[j]; }
    float inv = 1.f / s;
    v8h h0, h1, h2, h3;
    for (int j = 0; j < 8; ++j) {
      h0[j] = (_Float16)(v[j]      * inv);
      h1[j] = (_Float16)(v[8 + j]  * inv);
      h2[j] = (_Float16)(v[16 + j] * inv);
      h3[j] = (_Float16)(v[24 + j] * inv);
    }
    *(v8h*)&sfrag[h][m][0]      = h0;
    *(v8h*)&sfrag[h][m + 16][0] = h1;
    *(v8h*)&sfrag[h][m][8]      = h2;
    *(v8h*)&sfrag[h][m + 16][8] = h3;
  }
  __syncthreads();
  int w = tid >> 5, lane = tid & 31, l4 = lane & 15, g = lane >> 4;
  v16h aS = *(const v16h*)&sfrag[w][lane][0];
  const _Float16* ap = attB + (((size_t)b * 3 + br) * 8 + w) * 1024;
  v16h b0 = *(const v16h*)(ap + lane * 16);
  v16h b1 = *(const v16h*)(ap + (32 + lane) * 16);
  v8f y0 = {}, y1 = {};
  y0 = wmma_(aS, b0, y0);
  y1 = wmma_(aS, b1, y1);
  // write y (f16) into sfrag[w] as A-frag over this head's 32 columns (kt=w)
  for (int lt = 0; lt < 2; ++lt) {
    for (int r = 0; r < 8; ++r) {
      float yv = lt ? y1[r] : y0[r];
      int lanep = (r + 8 * g) + 16 * ((l4 >> 3) & 1);
      int e = 8 * lt + (l4 & 7);
      sfrag[w][lanep][e] = (_Float16)yv;
    }
  }
  __syncthreads();
  v8f z0 = {}, z1 = {};
  for (int kt = 0; kt < 8; ++kt) {
    v16h a = *(const v16h*)&sfrag[kt][lane][0];
    v16h w0 = *(const v16h*)(WS + (size_t)projSlot * 65536 + (((2 * w) * 8 + kt) * 32 + lane) * 16);
    v16h w1 = *(const v16h*)(WS + (size_t)projSlot * 65536 + (((2 * w + 1) * 8 + kt) * 32 + lane) * 16);
    z0 = wmma_(a, w0, z0);
    z1 = wmma_(a, w1, z1);
  }
  for (int cl = 0; cl < 2; ++cl) {
    int col = 32 * w + 16 * cl + l4;
    float bias = FB[projSlot * 256 + col];
    for (int r = 0; r < 8; ++r) {
      size_t addr = ((size_t)b * T_ + rt * 16 + r + 8 * g) * 256 + col;
      float v = (cl ? z1[r] : z0[r]) + bias;
      if (mode == 0) outb[addr] += v;
      else           outb[addr] = v;
    }
  }
}

// ---------------------------------------------------------------------------
// Kernel 6: per-batch scale/shift = silu(other_emb) @ emb_w^T + emb_b
// ---------------------------------------------------------------------------
__global__ __launch_bounds__(256) void style_emb(
    const float* __restrict__ other, const float* __restrict__ ew,
    const float* __restrict__ eb, float* __restrict__ ssb)
{
  int b = blockIdx.x, tid = threadIdx.x;
  __shared__ float se[TE_];
  for (int i = tid; i < TE_; i += 256) {
    float v = other[(size_t)b * TE_ + i];
    se[i] = v / (1.f + __expf(-v));
  }
  __syncthreads();
  for (int o = tid; o < 512; o += 256) {
    const float* wr = ew + (size_t)o * TE_;
    float a = eb[o];
    for (int k = 0; k < TE_; ++k) a += wr[k] * se[k];
    ssb[(size_t)b * 512 + o] = a;
  }
}

// ---------------------------------------------------------------------------
// Kernel 7: stylization out: h = silu(LN(q2)*(1+scale)+shift) @ out_w^T + out_b
// out = x + h.  grid (B, T/16).
// ---------------------------------------------------------------------------
__global__ __launch_bounds__(256) void style_out(
    const float* __restrict__ q2, const float* __restrict__ ssb,
    const float* __restrict__ sg, const float* __restrict__ sb,
    const _Float16* __restrict__ WS, const float* __restrict__ FB,
    const float* __restrict__ x, float* __restrict__ out)
{
  int b = blockIdx.x, rt = blockIdx.y, tid = threadIdx.x;
  __shared__ float red[2][16][16];
  __shared__ float mrs[16][2];
  __shared__ alignas(32) _Float16 hfrag[8][32][16];
  int m = tid >> 4, cs = tid & 15;
  const float* qp = q2 + ((size_t)b * T_ + rt * 16 + m) * 256 + cs * 16;
  float v[16]; float s = 0.f, s2 = 0.f;
  for (int j = 0; j < 16; ++j) { float t = qp[j]; v[j] = t; s += t; s2 += t * t; }
  red[0][m][cs] = s; red[1][m][cs] = s2;
  __syncthreads();
  if (cs == 0) {
    float a = 0.f, a2 = 0.f;
    for (int i = 0; i < 16; ++i) { a += red[0][m][i]; a2 += red[1][m][i]; }
    float mean = a / 256.f;
    float var = a2 / 256.f - mean * mean;
    mrs[m][0] = mean; mrs[m][1] = rsqrtf(var + EPS_);
  }
  __syncthreads();
  float mean = mrs[m][0], rstd = mrs[m][1];
  v8h lo, hi;
  for (int j = 0; j < 16; ++j) {
    int c = cs * 16 + j;
    float hn = (v[j] - mean) * rstd * sg[c] + sb[c];
    float sc = ssb[(size_t)b * 512 + c];
    float sh = ssb[(size_t)b * 512 + 256 + c];
    float hh = hn * (1.f + sc) + sh;
    float a = hh / (1.f + __expf(-hh)); // silu
    if (j < 8) lo[j] = (_Float16)a; else hi[j - 8] = (_Float16)a;
  }
  int kt = cs >> 1, eo = 8 * (cs & 1);
  *(v8h*)&hfrag[kt][m][eo] = lo;
  *(v8h*)&hfrag[kt][m + 16][eo] = hi;
  __syncthreads();
  int w = tid >> 5, lane = tid & 31, l4 = lane & 15, g = lane >> 4;
  v8f z0 = {}, z1 = {};
  for (int k = 0; k < 8; ++k) {
    v16h a = *(const v16h*)&hfrag[k][lane][0];
    v16h w0 = *(const v16h*)(WS + (size_t)SL_OUT * 65536 + (((2 * w) * 8 + k) * 32 + lane) * 16);
    v16h w1 = *(const v16h*)(WS + (size_t)SL_OUT * 65536 + (((2 * w + 1) * 8 + k) * 32 + lane) * 16);
    z0 = wmma_(a, w0, z0);
    z1 = wmma_(a, w1, z1);
  }
  for (int cl = 0; cl < 2; ++cl) {
    int col = 32 * w + 16 * cl + l4;
    float bias = FB[SL_OUT * 256 + col];
    for (int r = 0; r < 8; ++r) {
      size_t addr = ((size_t)b * T_ + rt * 16 + r + 8 * g) * 256 + col;
      out[addr] = x[addr] + (cl ? z1[r] : z0[r]) + bias;
    }
  }
}

// ---------------------------------------------------------------------------
extern "C" void kernel_launch(void* const* d_in, const int* in_sizes, int n_in,
                              void* d_out, int out_size, void* d_ws, size_t ws_size,
                              hipStream_t stream)
{
  (void)in_sizes; (void)n_in; (void)out_size; (void)ws_size;
  const float* x          = (const float*)d_in[0];
  const float* text_emb   = (const float*)d_in[1];
  const float* joint_emb  = (const float*)d_in[2];
  const float* other_emb  = (const float*)d_in[3];
  const float* src_mask   = (const float*)d_in[4];
  const float* joint_mask = (const float*)d_in[5];
  const int* pnt = (const int*)d_in[52];
  const int* pnb = (const int*)d_in[53];
  const int* pns = (const int*)d_in[54];

  char* wsb = (char*)d_ws;
  size_t off = 0;
  auto take = [&](size_t bytes) -> char* {
    char* p = wsb + off;
    off = (off + bytes + 255) & ~(size_t)255;
    return p;
  };
  _Float16* WS  = (_Float16*)take((size_t)NSLOT * 65536 * 2);
  float*    FB  = (float*)   take((size_t)NSLOT * 256 * 4);
  _Float16* xA  = (_Float16*)take((size_t)B_ * 64 * 8 * 512 * 2);
  _Float16* tcA = (_Float16*)take((size_t)B_ * 5 * 8 * 512 * 2);
  _Float16* jcA = (_Float16*)take((size_t)B_ * 2 * 2 * 512 * 2);
  float* q0   = (float*)take((size_t)B_ * T_ * 256 * 4);
  float* qf   = (float*)take((size_t)B_ * T_ * 256 * 4);
  float* q2   = (float*)take((size_t)B_ * T_ * 256 * 4);
  float* patt = (float*)take((size_t)B_ * 3 * 5 * 8 * 1024 * 4);
  float* pS   = (float*)take((size_t)B_ * 3 * 5 * 8 * 64 * 4);
  _Float16* attB = (_Float16*)take((size_t)B_ * 3 * 8 * 1024 * 2);
  float* ssb  = (float*)take((size_t)B_ * 512 * 4);

  FuseArgs fa;
  auto set = [&](int i, int wi, int gi, int bvi, int bi, int Dc) {
    fa.d[i].W  = (const float*)d_in[wi];
    fa.d[i].g  = gi  < 0 ? nullptr : (const float*)d_in[gi];
    fa.d[i].bv = bvi < 0 ? nullptr : (const float*)d_in[bvi];
    fa.d[i].b  = bi  < 0 ? nullptr : (const float*)d_in[bi];
    fa.d[i].Dc = Dc;
  };
  set(SL_Q,     8,  6,  7,  9, 256);
  set(SL_TKX,  18, 10, 11, 19, 256);
  set(SL_TVX,  20, 10, 11, 21, 256);
  set(SL_TKC,  14, 12, 13, 15, 256);
  set(SL_TVC,  16, 12, 13, 17, 256);
  set(SL_TPROJ,22, -1, -1, 23, 256);
  set(SL_JKX,  32, 24, 25, 33, 256);
  set(SL_JVX,  34, 24, 25, 35, 256);
  set(SL_JKC,  28, 26, 27, 29,  64);
  set(SL_JVC,  30, 26, 27, 31,  64);
  set(SL_JPROJ,36, -1, -1, 37, 256);
  set(SL_YKX,  40, 38, 39, 41, 256);
  set(SL_YVX,  42, 38, 39, 43, 256);
  set(SL_YPROJ,44, -1, -1, 45, 256);
  set(SL_OUT,  50, -1, -1, 51, 256);

  fuse_weights<<<dim3(NSLOT), dim3(256), 0, stream>>>(fa, WS, FB);
  style_emb<<<dim3(B_), dim3(256), 0, stream>>>(other_emb,
      (const float*)d_in[46], (const float*)d_in[47], ssb);
  prep_ln<<<dim3(B_, 64), dim3(256), 0, stream>>>(x, xA, 1024, 64, 256);
  prep_ln<<<dim3(B_, 5),  dim3(256), 0, stream>>>(text_emb, tcA, 77, 5, 256);
  prep_ln<<<dim3(B_, 2),  dim3(256), 0, stream>>>(joint_emb, jcA, 32, 2, 64);
  q_gemm<<<dim3(B_, 64), dim3(256), 0, stream>>>(xA, WS, FB, q0, qf);
  attn_build<<<dim3(B_, 3, 5), dim3(256), 0, stream>>>(
      xA, tcA, jcA, WS, FB, src_mask, joint_mask, pnt, pnb, pns, patt, pS);
  finalize_att<<<dim3(B_, 3), dim3(256), 0, stream>>>(patt, pS, attB, pnt, pnb, pns);
  y_apply<<<dim3(B_, 64), dim3(256), 0, stream>>>(q0, attB, WS, FB, SL_TPROJ, 0, 0, qf, pnt, pnb, pns);
  y_apply<<<dim3(B_, 64), dim3(256), 0, stream>>>(q0, attB, WS, FB, SL_JPROJ, 1, 0, qf, pnt, pnb, pns);
  y_apply<<<dim3(B_, 64), dim3(256), 0, stream>>>(qf, attB, WS, FB, SL_YPROJ, 2, 1, q2, pnt, pnb, pns);
  style_out<<<dim3(B_, 64), dim3(256), 0, stream>>>(q2, ssb,
      (const float*)d_in[48], (const float*)d_in[49], WS, FB, x, (float*)d_out);
}
